// MultiFitAttention_52759378264631
// MI455X (gfx1250) — compile-verified, matrix operand order checked by reading, not run on hardware
//
#include <hip/hip_runtime.h>
#include <hip/hip_bf16.h>
#include <math.h>

// ---------------------------------------------------------------------------
// Types for WMMA fragments
// ---------------------------------------------------------------------------
typedef __attribute__((ext_vector_type(16))) __bf16          bf16x16;
typedef __attribute__((ext_vector_type(8)))  float           floatx8;
typedef __attribute__((ext_vector_type(8)))  unsigned short  ushortx8;
typedef __attribute__((ext_vector_type(16))) unsigned short  ushortx16;

#define B_  16
#define S_  4096
#define H_  512
#define K_  1024              // 2*H : reduce dim of enc @ W2^T
#define SROWS 64              // rows of enc staged per workgroup
#define LDS_ROW_BYTES 2064    // 1024 bf16 = 2048 B + 16 B pad (4-dword bank skew)

__device__ __forceinline__ unsigned short f2bf(float x) {
    unsigned u = __builtin_bit_cast(unsigned, x);
    u += 0x7fffu + ((u >> 16) & 1u);          // round-to-nearest-even
    return (unsigned short)(u >> 16);
}

__device__ __forceinline__ bf16x16 make_frag(ushortx8 lo, ushortx8 hi) {
    ushortx16 t;
#pragma unroll
    for (int i = 0; i < 8; ++i) { t[i] = lo[i]; t[i + 8] = hi[i]; }
    return __builtin_bit_cast(bf16x16, t);
}

__device__ __forceinline__ float fast_tanh(float x) {
#if defined(__has_builtin) && __has_builtin(__builtin_amdgcn_tanhf)
    return __builtin_amdgcn_tanhf(x);        // native v_tanh_f32 (TRANS op)
#else
    float r;
    asm("v_tanh_f32 %0, %1" : "=v"(r) : "v"(x));
    return r;
#endif
}

__device__ __forceinline__ floatx8 wmma_bf16(bf16x16 a, bf16x16 b, floatx8 c) {
    return __builtin_amdgcn_wmma_f32_16x16x32_bf16(
        false, a, false, b, (short)0, c, false, false);
}

// ---------------------------------------------------------------------------
// Prep 1: W2 (= W_attn[:, H:3H]) f32 -> bf16, row-major [H_, K_]
// ---------------------------------------------------------------------------
__global__ __launch_bounds__(256)
void mfa_prep_w2(const float* __restrict__ W, unsigned short* __restrict__ w2bf) {
    int t  = blockIdx.x * 256 + threadIdx.x;     // 131072 threads, one float4 each
    int h  = t >> 8;                             // 0..511
    int c4 = t & 255;                            // 0..255
    float4 f = *(const float4*)(W + (size_t)h * (3 * H_) + H_ + c4 * 4);
    ushort4 o;
    o.x = f2bf(f.x); o.y = f2bf(f.y); o.z = f2bf(f.z); o.w = f2bf(f.w);
    *(ushort4*)(w2bf + (size_t)h * K_ + c4 * 4) = o;
}

// ---------------------------------------------------------------------------
// Prep 2: part1[b,h] = hidden[b,:] . W1[h,:] + bias[h]   (f32, tiny)
// ---------------------------------------------------------------------------
__global__ __launch_bounds__(256)
void mfa_prep_part1(const float* __restrict__ hidden, const float* __restrict__ W,
                    const float* __restrict__ bias, float* __restrict__ part1) {
    int t = blockIdx.x * 256 + threadIdx.x;      // 8192 threads
    int b = t >> 9;
    int h = t & (H_ - 1);
    const float* hr = hidden + (size_t)b * H_;
    const float* wr = W + (size_t)h * (3 * H_);
    float s = 0.f;
#pragma unroll 4
    for (int f = 0; f < H_; f += 4) {
        float4 a = *(const float4*)(hr + f);
        float4 w = *(const float4*)(wr + f);
        s += a.x * w.x + a.y * w.y + a.z * w.z + a.w * w.w;
    }
    part1[t] = s + bias[h];
}

// ---------------------------------------------------------------------------
// Main: logits[b,s] = v . tanh( enc[b,s,:] @ W2^T + part1[b,:] )
// Block = 4 waves = 64 enc rows for one batch. Enc strip staged once to LDS
// as bf16. Each wave: 16-row strip, N blocked by 2 tiles (A-frag reuse),
// K pipelined with explicit double buffering, fused v_tanh_f32 epilogue.
// ---------------------------------------------------------------------------
__global__ __launch_bounds__(128)
void mfa_logits(const float* __restrict__ enc,          // [B, S, K_] f32
                const unsigned short* __restrict__ w2,  // [H_, K_] bf16
                const float* __restrict__ part1,        // [B, H_]
                const float* __restrict__ vvec,         // [H_]
                float* __restrict__ logits) {           // [B, S]
    extern __shared__ char smem[];
    const int b   = blockIdx.y;
    const int s0  = blockIdx.x * SROWS;
    const int tid = threadIdx.x;

    // ---- Stage 64 x 1024 f32 -> bf16 into LDS (read enc exactly once) ----
    const float* src = enc + ((size_t)b * S_ + s0) * K_;
#pragma unroll 4
    for (int i = tid; i < SROWS * 256; i += 128) {   // 256 float4 per row
        int row = i >> 8, c4 = i & 255;
        float4 f = *(const float4*)(src + (size_t)row * K_ + c4 * 4);
        ushort4 hvec;
        hvec.x = f2bf(f.x); hvec.y = f2bf(f.y); hvec.z = f2bf(f.z); hvec.w = f2bf(f.w);
        *(ushort4*)(smem + row * LDS_ROW_BYTES + c4 * 8) = hvec;
    }
    __syncthreads();

    const int wave = tid >> 5;        // 0..3 -> 16-row sub-strip
    const int lane = tid & 31;
    const int lrow = lane & 15;       // M row within tile / N col within tile
    const int half = lane >> 4;       // K-phase per ISA 16-bit fragment layout

    const char* aPtr = smem + (wave * 16 + lrow) * LDS_ROW_BYTES + half * 16;
    float rowacc[8] = {0.f, 0.f, 0.f, 0.f, 0.f, 0.f, 0.f, 0.f};

    for (int ntb = 0; ntb < H_ / 16; ntb += 2) {  // 16 iterations, 2 N-tiles each
        const int n0 = ntb * 16 + lrow;
        const int n1 = n0 + 16;
        const unsigned short* bRow0 = w2 + (size_t)n0 * K_ + half * 8;
        const unsigned short* bRow1 = w2 + (size_t)n1 * K_ + half * 8;

        // epilogue scalars issued early so they are in flight during the GEMM
        const float p0  = part1[b * H_ + n0];
        const float p1  = part1[b * H_ + n1];
        const float vn0 = vvec[n0];
        const float vn1 = vvec[n1];

        floatx8 acc0 = (floatx8)0.0f;
        floatx8 acc1 = (floatx8)0.0f;

        // ---- prologue: fragments for k0 = 0 ----
        ushortx8 alo  = *(const ushortx8*)(aPtr);
        ushortx8 ahi  = *(const ushortx8*)(aPtr + 32);
        ushortx8 b0lo = *(const ushortx8*)(bRow0);
        ushortx8 b0hi = *(const ushortx8*)(bRow0 + 16);
        ushortx8 b1lo = *(const ushortx8*)(bRow1);
        ushortx8 b1hi = *(const ushortx8*)(bRow1 + 16);

#pragma unroll 2
        for (int k0 = 32; k0 < K_; k0 += 32) {
            // prefetch next K step while current WMMAs execute
            ushortx8 alo_n  = *(const ushortx8*)(aPtr + k0 * 2);
            ushortx8 ahi_n  = *(const ushortx8*)(aPtr + k0 * 2 + 32);
            ushortx8 b0lo_n = *(const ushortx8*)(bRow0 + k0);
            ushortx8 b0hi_n = *(const ushortx8*)(bRow0 + k0 + 16);
            ushortx8 b1lo_n = *(const ushortx8*)(bRow1 + k0);
            ushortx8 b1hi_n = *(const ushortx8*)(bRow1 + k0 + 16);

            bf16x16 a = make_frag(alo, ahi);
            acc0 = wmma_bf16(a, make_frag(b0lo, b0hi), acc0);
            acc1 = wmma_bf16(a, make_frag(b1lo, b1hi), acc1);

            alo = alo_n; ahi = ahi_n;
            b0lo = b0lo_n; b0hi = b0hi_n;
            b1lo = b1lo_n; b1hi = b1hi_n;
        }
        {   // drain
            bf16x16 a = make_frag(alo, ahi);
            acc0 = wmma_bf16(a, make_frag(b0lo, b0hi), acc0);
            acc1 = wmma_bf16(a, make_frag(b1lo, b1hi), acc1);
        }

        // ---- fused epilogue: v[n] * tanh(score + part1) ----
#pragma unroll
        for (int r = 0; r < 8; ++r) {
            float t0 = fast_tanh(acc0[r] + p0);
            float t1 = fast_tanh(acc1[r] + p1);
            rowacc[r] += vn0 * t0 + vn1 * t1;
        }
    }

    // Reduce over the 16 lanes of each half (each half owns distinct M rows)
#pragma unroll
    for (int r = 0; r < 8; ++r) {
        float x = rowacc[r];
        x += __shfl_xor(x, 1, 32);
        x += __shfl_xor(x, 2, 32);
        x += __shfl_xor(x, 4, 32);
        x += __shfl_xor(x, 8, 32);
        rowacc[r] = x;
    }
    if (lrow == 0) {
        float* dst = logits + (size_t)b * S_ + s0 + wave * 16 + half * 8;
#pragma unroll
        for (int r = 0; r < 8; ++r) dst[r] = rowacc[r];
    }
}

// ---------------------------------------------------------------------------
// Softmax over S per batch row
// ---------------------------------------------------------------------------
__global__ __launch_bounds__(256)
void mfa_softmax(const float* __restrict__ logits, float* __restrict__ out) {
    __shared__ float red[256];
    const int b = blockIdx.x;
    const int t = threadIdx.x;
    const float* row = logits + (size_t)b * S_;

    float m = -3.402823466e+38f;
    for (int i = t; i < S_; i += 256) m = fmaxf(m, row[i]);
    red[t] = m; __syncthreads();
    for (int s = 128; s > 0; s >>= 1) {
        if (t < s) red[t] = fmaxf(red[t], red[t + s]);
        __syncthreads();
    }
    m = red[0]; __syncthreads();

    float sum = 0.f;
    for (int i = t; i < S_; i += 256) sum += __expf(row[i] - m);
    red[t] = sum; __syncthreads();
    for (int s = 128; s > 0; s >>= 1) {
        if (t < s) red[t] += red[t + s];
        __syncthreads();
    }
    const float inv = 1.0f / red[0];

    for (int i = t; i < S_; i += 256)
        out[(size_t)b * S_ + i] = __expf(row[i] - m) * inv;
}

// ---------------------------------------------------------------------------
// Launch
// ---------------------------------------------------------------------------
extern "C" void kernel_launch(void* const* d_in, const int* in_sizes, int n_in,
                              void* d_out, int out_size, void* d_ws, size_t ws_size,
                              hipStream_t stream) {
    const float* hidden = (const float*)d_in[0];   // [16, 512]
    const float* enc    = (const float*)d_in[1];   // [16, 4096, 1024]
    const float* W      = (const float*)d_in[2];   // [512, 1536]
    const float* bias   = (const float*)d_in[3];   // [512]
    const float* vvec   = (const float*)d_in[4];   // [512]
    float*       out    = (float*)d_out;           // [16, 4096]

    char* ws = (char*)d_ws;
    unsigned short* w2bf  = (unsigned short*)ws;                       // 1 MB
    float*          part1 = (float*)(ws + 1048576);                    // 32 KB
    float*          logit = (float*)(ws + 1048576 + 32768);            // 256 KB

    // Allow 129 KB of dynamic LDS (CDNA5 WGP has 320 KB); ignore error if capped.
    (void)hipFuncSetAttribute((const void*)mfa_logits,
                              hipFuncAttributeMaxDynamicSharedMemorySize,
                              SROWS * LDS_ROW_BYTES);

    mfa_prep_w2   <<<512, 256, 0, stream>>>(W, w2bf);
    mfa_prep_part1<<<32,  256, 0, stream>>>(hidden, W, bias, part1);

    dim3 grid(S_ / SROWS, B_);   // 64 x 16 workgroups
    mfa_logits<<<grid, 128, SROWS * LDS_ROW_BYTES, stream>>>(enc, w2bf, part1, vvec, logit);

    mfa_softmax<<<B_, 256, 0, stream>>>(logit, out);
}
